// AudioEncoderAttention_78537771975264
// MI455X (gfx1250) — matrix operand order repaired and checked by reading, not compiled, not run on hardware
//
#include <hip/hip_runtime.h>

typedef __bf16 bf16_t;
typedef __attribute__((ext_vector_type(16))) __bf16 v16bf;
typedef __attribute__((ext_vector_type(8)))  __bf16 v8bf;
typedef __attribute__((ext_vector_type(8)))  float   v8f;

#define EDIM 1024
#define HHEADS 16
#define DHEAD 64
#define NSEQ 8
#define SEQL 1024
#define TTOK (NSEQ * SEQL)

union V16 { v16bf v; v8bf h[2]; };

__device__ __forceinline__ v8f wmma_bf16(v16bf a, v16bf b, v8f c) {
  // D = A(16x32 bf16) x B(32x16 bf16) + C(16x16 f32)
  return __builtin_amdgcn_wmma_f32_16x16x32_bf16(false, a, false, b, (short)0, c,
                                                 false, false);
}

// ---- async global -> LDS copy helpers (ASYNCcnt-tracked, wave-level) ----
__device__ __forceinline__ unsigned lds_off(const void* p) {
  return (unsigned)(size_t)p; // LDS aperture: low 32 bits = LDS offset
}
__device__ __forceinline__ void async_copy16(unsigned lds_dst, const bf16_t* gsrc) {
  asm volatile("global_load_async_to_lds_b128 %0, %1, off"
               :: "v"(lds_dst), "v"(gsrc) : "memory");
}
#define WAIT_ASYNC(n) asm volatile("s_wait_asynccnt " #n ::: "memory")

// ---------------------------------------------------------------- cast f32->bf16
__global__ __launch_bounds__(256) void cast_kernel(const float* __restrict__ in,
                                                   bf16_t* __restrict__ out, int n) {
  int i = blockIdx.x * blockDim.x + threadIdx.x;
  if (i < n) out[i] = (bf16_t)in[i];
}

// ---------------------------------------------------------------- shared GEMM core
// Workgroup: 8 waves, computes 128 rows x 64 cols. Weight tile (64x32 bf16 = 4KB)
// staged in LDS via double-buffered async copies; last K-step peeled so the
// steady-state loop is branch-free (keeps WMMA accumulators pinned).
__device__ __forceinline__ void gemm_acc_128x64(
    const bf16_t* __restrict__ X, const bf16_t* __restrict__ W,
    int tblock, int ebase, bf16_t (*wtile)[64 * 32], v8f acc[4]) {
  const int lane = threadIdx.x & 31;
  const int wave = threadIdx.x >> 5;
  const int n = lane & 15;
  const int g = lane >> 4;
  const int tid = threadIdx.x;
  const int wrow = tid >> 2;          // 0..63 : weight row (output column)
  const int wchunk = (tid & 3) * 8;   // 16B chunk within 32-wide K slice
  const bf16_t* wsrc = W + (size_t)(ebase + wrow) * EDIM + wchunk;
  const unsigned dst0 = lds_off(&wtile[0][wrow * 32 + wchunk]);
  const unsigned dst1 = lds_off(&wtile[1][wrow * 32 + wchunk]);
  const bf16_t* arow = X + (size_t)(tblock + wave * 16 + n) * EDIM;

  auto step = [&](int i) {
    const int kb = i * 32;
    V16 a;
    a.h[0] = *(const v8bf*)(arow + kb + g * 16);
    a.h[1] = *(const v8bf*)(arow + kb + g * 16 + 8);
    const bf16_t* bt = wtile[i & 1];
#pragma unroll
    for (int f = 0; f < 4; ++f) {
      v16bf b = *(const v16bf*)(bt + (f * 16 + n) * 32 + g * 16);
      acc[f] = wmma_bf16(a.v, b, acc[f]);
    }
  };

  async_copy16(dst0, wsrc); // kb = 0
  const int nIter = EDIM / 32;
  for (int i = 0; i < nIter - 1; ++i) {
    async_copy16((i & 1) ? dst0 : dst1, wsrc + i * 32 + 32);
    WAIT_ASYNC(0x1); // in-order completion: current buffer resident
    __syncthreads();
    step(i);
    __syncthreads(); // all waves done reading before next overwrite
  }
  WAIT_ASYNC(0x0);
  __syncthreads();
  step(nIter - 1);
}

// ---------------------------------------------------------------- QKV projection + bias + RoPE
// grid: (TTOK/128, EDIM/64, 3)  block: 256
__global__ __launch_bounds__(256) void qkv_kernel(
    const bf16_t* __restrict__ hs,
    const bf16_t* __restrict__ wq, const bf16_t* __restrict__ wk,
    const bf16_t* __restrict__ wv,
    const float* __restrict__ bq, const float* __restrict__ bv,
    const float* __restrict__ cosp, const float* __restrict__ sinp,
    bf16_t* __restrict__ qout, bf16_t* __restrict__ kout,
    bf16_t* __restrict__ vtout) {
  const int lane = threadIdx.x & 31;
  const int wave = threadIdx.x >> 5;
  const int n = lane & 15;
  const int g = lane >> 4;
  const int tblock = blockIdx.x * 128;
  const int tbase = tblock + wave * 16;
  const int head  = blockIdx.y;
  const int ebase = head * 64;
  const int which = blockIdx.z; // 0=Q 1=K 2=V

  __shared__ alignas(16) bf16_t wtile[2][64 * 32];
  const bf16_t* w = (which == 0) ? wq : (which == 1) ? wk : wv;

  v8f acc[4] = {};
  gemm_acc_128x64(hs, w, tblock, ebase, wtile, acc);

  // bias (Q: bq, V: bv, K: none)
  if (which != 1) {
    const float* bias = (which == 0) ? bq : bv;
#pragma unroll
    for (int f = 0; f < 4; ++f) {
      float bb = bias[ebase + f * 16 + n];
#pragma unroll
      for (int j = 0; j < 8; ++j) acc[f][j] += bb;
    }
  }

  if (which < 2) {
    // RoPE: partner d +/- 32 lives in same lane, fragment f <-> f+2
#pragma unroll
    for (int f = 0; f < 2; ++f) {
      int d1 = f * 16 + n;
#pragma unroll
      for (int j = 0; j < 8; ++j) {
        int t = tbase + 8 * g + j;
        float cv1 = cosp[t * DHEAD + d1],      sv1 = sinp[t * DHEAD + d1];
        float cv2 = cosp[t * DHEAD + d1 + 32], sv2 = sinp[t * DHEAD + d1 + 32];
        float x1 = acc[f][j], x2 = acc[f + 2][j];
        acc[f][j]     = x1 * cv1 - x2 * sv1;
        acc[f + 2][j] = x2 * cv2 + x1 * sv2;
      }
    }
    bf16_t* o = (which == 0) ? qout : kout;
#pragma unroll
    for (int f = 0; f < 4; ++f)
#pragma unroll
      for (int j = 0; j < 8; ++j)
        o[(size_t)(tbase + 8 * g + j) * EDIM + ebase + f * 16 + n] = (bf16_t)acc[f][j];
  } else {
    // V stored transposed: Vt[seq][head][d][t_local] -> 16B packed store per frag
    int seq = tbase / SEQL;
    int tl  = tbase % SEQL;
#pragma unroll
    for (int f = 0; f < 4; ++f) {
      int d = f * 16 + n;
      union { bf16_t b[8]; v8bf v; } pk;
#pragma unroll
      for (int j = 0; j < 8; ++j) pk.b[j] = (bf16_t)acc[f][j];
      size_t idx = (((size_t)seq * HHEADS + head) * DHEAD + d) * SEQL + tl + 8 * g;
      *(v8bf*)(vtout + idx) = pk.v;
    }
  }
}

// ---------------------------------------------------------------- flash attention per (seq, head)
// grid: (SEQL/128, HHEADS, NSEQ)  block: 256; each wave owns 16 query rows.
// K and V tiles (4KB each) staged in LDS via double-buffered async copies;
// last t-step peeled so the steady-state loop is branch-free.
__global__ __launch_bounds__(256) void attn_kernel(
    const bf16_t* __restrict__ Q, const bf16_t* __restrict__ K,
    const bf16_t* __restrict__ Vt, bf16_t* __restrict__ AO) {
  const int lane = threadIdx.x & 31;
  const int wave = threadIdx.x >> 5;
  const int n = lane & 15;
  const int g = lane >> 4;
  const int head = blockIdx.y;
  const int seq  = blockIdx.z;
  const int rbase = blockIdx.x * 128 + wave * 16;

  __shared__ alignas(16) bf16_t Ktile[2][32 * 64]; // [t_rel][d]
  __shared__ alignas(16) bf16_t Vtile[2][64 * 32]; // [d][t_rel]
  __shared__ alignas(32) bf16_t pbuf[8][16 * 32];  // per-wave P transpose buffer
  bf16_t* pb = pbuf[wave];

  // Q A-fragments (d 0..31, 32..63) live in registers for the whole kernel
  const bf16_t* qrow = Q + (size_t)(seq * SEQL + rbase + n) * EDIM + head * 64;
  V16 aq0, aq1;
  aq0.h[0] = *(const v8bf*)(qrow + g * 16);
  aq0.h[1] = *(const v8bf*)(qrow + g * 16 + 8);
  aq1.h[0] = *(const v8bf*)(qrow + 32 + g * 16);
  aq1.h[1] = *(const v8bf*)(qrow + 40 + g * 16);

  v8f o0 = {}, o1 = {}, o2 = {}, o3 = {};
  float mi[8], li[8];
#pragma unroll
  for (int j = 0; j < 8; ++j) { mi[j] = -1e30f; li[j] = 0.f; }

  // async staging assignments: 256 threads x 16B = 4KB tile
  const int tid = threadIdx.x;
  const int ktrow = tid >> 3, kchunk = (tid & 7) * 8;
  const int vdrow = tid >> 2, vchunk = (tid & 3) * 8;
  const bf16_t* ksrc = K + (size_t)(seq * SEQL + ktrow) * EDIM + head * 64 + kchunk;
  const bf16_t* vsrc = Vt + ((size_t)seq * HHEADS + head) * DHEAD * SEQL +
                       (size_t)vdrow * SEQL + vchunk;
  const unsigned kd0 = lds_off(&Ktile[0][ktrow * 64 + kchunk]);
  const unsigned kd1 = lds_off(&Ktile[1][ktrow * 64 + kchunk]);
  const unsigned vd0 = lds_off(&Vtile[0][vdrow * 32 + vchunk]);
  const unsigned vd1 = lds_off(&Vtile[1][vdrow * 32 + vchunk]);

  const float scale = 0.125f; // 1/sqrt(64)

  auto step = [&](int i) {
    const bf16_t* kt = Ktile[i & 1];
    const bf16_t* vt = Vtile[i & 1];

    // S = Q K^T for 32 keys (two 16-wide column tiles) from LDS
    v8f s0 = {}, s1 = {};
    s0 = wmma_bf16(aq0.v, *(const v16bf*)(kt + n * 64 + g * 16), s0);
    s0 = wmma_bf16(aq1.v, *(const v16bf*)(kt + n * 64 + 32 + g * 16), s0);
    s1 = wmma_bf16(aq0.v, *(const v16bf*)(kt + (16 + n) * 64 + g * 16), s1);
    s1 = wmma_bf16(aq1.v, *(const v16bf*)(kt + (16 + n) * 64 + 32 + g * 16), s1);

    // online softmax; row (8g+j) stats live in the 16-lane half that holds it
#pragma unroll
    for (int j = 0; j < 8; ++j) {
      float a = s0[j] * scale, b = s1[j] * scale;
      float mx = fmaxf(a, b);
#pragma unroll
      for (int m = 1; m <= 8; m <<= 1) mx = fmaxf(mx, __shfl_xor(mx, m, 32));
      float mnew = fmaxf(mi[j], mx);
      float corr = __expf(mi[j] - mnew);
      float p0 = __expf(a - mnew), p1 = __expf(b - mnew);
      float rs = p0 + p1;
#pragma unroll
      for (int m = 1; m <= 8; m <<= 1) rs += __shfl_xor(rs, m, 32);
      li[j] = li[j] * corr + rs;
      mi[j] = mnew;
      o0[j] *= corr; o1[j] *= corr; o2[j] *= corr; o3[j] *= corr;
      pb[(8 * g + j) * 32 + n]      = (bf16_t)p0;
      pb[(8 * g + j) * 32 + 16 + n] = (bf16_t)p1;
    }
    __syncthreads(); // pbuf ready
    V16 ap;
    ap.h[0] = *(const v8bf*)(pb + n * 32 + g * 16);
    ap.h[1] = *(const v8bf*)(pb + n * 32 + g * 16 + 8);
    // O += P * V from LDS (V transposed -> contiguous B fragments)
    o0 = wmma_bf16(ap.v, *(const v16bf*)(vt + (0 * 16 + n) * 32 + g * 16), o0);
    o1 = wmma_bf16(ap.v, *(const v16bf*)(vt + (1 * 16 + n) * 32 + g * 16), o1);
    o2 = wmma_bf16(ap.v, *(const v16bf*)(vt + (2 * 16 + n) * 32 + g * 16), o2);
    o3 = wmma_bf16(ap.v, *(const v16bf*)(vt + (3 * 16 + n) * 32 + g * 16), o3);
    __syncthreads(); // all reads done before next iteration's async overwrite
  };

  async_copy16(kd0, ksrc);
  async_copy16(vd0, vsrc);

  const int nT = SEQL / 32;
  for (int i = 0; i < nT - 1; ++i) {
    const int t0 = i * 32;
    async_copy16((i & 1) ? kd0 : kd1, ksrc + (size_t)(t0 + 32) * EDIM);
    async_copy16((i & 1) ? vd0 : vd1, vsrc + t0 + 32);
    WAIT_ASYNC(0x2); // two newest outstanding -> current K/V tiles resident
    __syncthreads();
    step(i);
  }
  WAIT_ASYNC(0x0);
  __syncthreads();
  step(nT - 1);

#pragma unroll
  for (int j = 0; j < 8; ++j) {
    float inv = 1.0f / li[j];
    size_t row = (size_t)(seq * SEQL + rbase + 8 * g + j) * EDIM + head * 64 + n;
    AO[row + 0]  = (bf16_t)(o0[j] * inv);
    AO[row + 16] = (bf16_t)(o1[j] * inv);
    AO[row + 32] = (bf16_t)(o2[j] * inv);
    AO[row + 48] = (bf16_t)(o3[j] * inv);
  }
}

// ---------------------------------------------------------------- output projection (fp32 out)
// grid: (TTOK/128, EDIM/64)  block: 256
__global__ __launch_bounds__(256) void proj_kernel(
    const bf16_t* __restrict__ X, const bf16_t* __restrict__ W,
    const float* __restrict__ bias, float* __restrict__ out) {
  const int lane = threadIdx.x & 31;
  const int wave = threadIdx.x >> 5;
  const int n = lane & 15;
  const int g = lane >> 4;
  const int tblock = blockIdx.x * 128;
  const int tbase = tblock + wave * 16;
  const int ebase = blockIdx.y * 64;

  __shared__ alignas(16) bf16_t wtile[2][64 * 32];
  v8f acc[4] = {};
  gemm_acc_128x64(X, W, tblock, ebase, wtile, acc);

#pragma unroll
  for (int f = 0; f < 4; ++f) {
    float bb = bias[ebase + f * 16 + n];
#pragma unroll
    for (int j = 0; j < 8; ++j)
      out[(size_t)(tbase + 8 * g + j) * EDIM + ebase + f * 16 + n] = acc[f][j] + bb;
  }
}

// ---------------------------------------------------------------- host launch
extern "C" void kernel_launch(void* const* d_in, const int* in_sizes, int n_in,
                              void* d_out, int out_size, void* d_ws, size_t ws_size,
                              hipStream_t stream) {
  const float* hs   = (const float*)d_in[0];
  const float* cosp = (const float*)d_in[1];
  const float* sinp = (const float*)d_in[2];
  const float* wq   = (const float*)d_in[3];
  const float* bq   = (const float*)d_in[4];
  const float* wk   = (const float*)d_in[5];
  const float* wv   = (const float*)d_in[6];
  const float* bv   = (const float*)d_in[7];
  const float* wo   = (const float*)d_in[8];
  const float* bo   = (const float*)d_in[9];

  bf16_t* hsb = (bf16_t*)d_ws;
  bf16_t* wqb = hsb + (size_t)TTOK * EDIM;
  bf16_t* wkb = wqb + (size_t)EDIM * EDIM;
  bf16_t* wvb = wkb + (size_t)EDIM * EDIM;
  bf16_t* wob = wvb + (size_t)EDIM * EDIM;
  bf16_t* Qb  = wob + (size_t)EDIM * EDIM;
  bf16_t* Kb  = Qb  + (size_t)TTOK * EDIM;
  bf16_t* Vtb = Kb  + (size_t)TTOK * EDIM;
  bf16_t* AOb = Vtb + (size_t)TTOK * EDIM;

  const int NHS = TTOK * EDIM;
  const int NW  = EDIM * EDIM;
  cast_kernel<<<(NHS + 255) / 256, 256, 0, stream>>>(hs, hsb, NHS);
  cast_kernel<<<(NW + 255) / 256, 256, 0, stream>>>(wq, wqb, NW);
  cast_kernel<<<(NW + 255) / 256, 256, 0, stream>>>(wk, wkb, NW);
  cast_kernel<<<(NW + 255) / 256, 256, 0, stream>>>(wv, wvb, NW);
  cast_kernel<<<(NW + 255) / 256, 256, 0, stream>>>(wo, wob, NW);

  qkv_kernel<<<dim3(TTOK / 128, EDIM / 64, 3), 256, 0, stream>>>(
      hsb, wqb, wkb, wvb, bq, bv, cosp, sinp, Qb, Kb, Vtb);

  attn_kernel<<<dim3(SEQL / 128, HHEADS, NSEQ), 256, 0, stream>>>(Qb, Kb, Vtb, AOb);

  proj_kernel<<<dim3(TTOK / 128, EDIM / 64), 256, 0, stream>>>(AOb, wob, bo,
                                                               (float*)d_out);
}
